// Dynamic_Directional_Attention_5437428596891
// MI455X (gfx1250) — matrix operand order repaired and checked by reading, not compiled, not run on hardware
//
#include <hip/hip_runtime.h>
#include <hip/hip_bf16.h>
#include <math.h>

typedef _Float16 v16h __attribute__((ext_vector_type(16)));
typedef _Float16 v8h  __attribute__((ext_vector_type(8)));
typedef float    v8f  __attribute__((ext_vector_type(8)));
typedef int      vi4  __attribute__((vector_size(16)));   // int4, matches builtin param

#define B_  4
#define L_  2048
#define S_  2048
#define H_  8
#define E_  64
#define EPS 1e-6f

#define KP  72    // LDS pitch (halves) for K tile rows: 36-word bank stride, conflict-free
#define VP  136   // LDS pitch (halves) for V tile rows: 68-word bank stride, conflict-free

union Frag { v16h v; v8h h[2]; };

#define AS1 __attribute__((address_space(1)))
#define AS3 __attribute__((address_space(3)))

#if defined(__has_builtin)
#  if __has_builtin(__builtin_amdgcn_global_load_async_to_lds_b128)
#    define HAVE_ASYNC_LDS 1
#  endif
#endif
#ifndef HAVE_ASYNC_LDS
#  define HAVE_ASYNC_LDS 0
#endif

// Stage one 16-byte chunk global -> LDS. Async (ASYNCcnt, no VGPR round trip)
// when the gfx1250 builtin is available; otherwise load + ds_store_b128.
__device__ __forceinline__ void stage16(const _Float16* g, _Float16* l) {
#if HAVE_ASYNC_LDS
  __builtin_amdgcn_global_load_async_to_lds_b128((AS1 vi4*)g, (AS3 vi4*)l, 0, 0);
#else
  *(v8h*)l = *(const v8h*)g;
#endif
}

__device__ __forceinline__ void stage_wait() {
#if HAVE_ASYNC_LDS
  asm volatile("s_wait_asynccnt 0x0" ::: "memory");
#endif
}

// ---------------------------------------------------------------------------
// Kernel 1: per-(b,h,e) inverse std (ddof=1) over the sequence dimension.
// ---------------------------------------------------------------------------
__global__ __launch_bounds__(64) void stats_kernel(
    const float* __restrict__ q, const float* __restrict__ k,
    float* __restrict__ inv_q, float* __restrict__ inv_k) {
  const int bh = blockIdx.x;
  const int b = bh >> 3, h = bh & 7;
  const int e = threadIdx.x;
  const float* qp = q + ((size_t)b * L_ * H_ + h) * E_ + e;
  const float* kp = k + ((size_t)b * S_ * H_ + h) * E_ + e;
  float qs = 0.f, qss = 0.f, ks = 0.f, kss = 0.f;
  for (int l = 0; l < L_; ++l) {
    float x = qp[(size_t)l * H_ * E_];
    qs += x; qss += x * x;
    float y = kp[(size_t)l * H_ * E_];
    ks += y; kss += y * y;
  }
  const float n = (float)L_;
  float vq = (qss - qs * qs / n) / (n - 1.0f);
  float vk = (kss - ks * ks / n) / (n - 1.0f);
  inv_q[bh * E_ + e] = 1.0f / (sqrtf(fmaxf(vq, 0.f)) + EPS);
  inv_k[bh * E_ + e] = 1.0f / (sqrtf(fmaxf(vk, 0.f)) + EPS);
}

// ---------------------------------------------------------------------------
// Kernel 2: transform Q/K -> f16 head-major [BH][L][E]; transpose V -> [BH][E][S].
// ---------------------------------------------------------------------------
__global__ __launch_bounds__(256) void transform_kernel(
    const float* __restrict__ q, const float* __restrict__ k,
    const float* __restrict__ v,
    const float* __restrict__ inv_q, const float* __restrict__ inv_k,
    const float* __restrict__ dwp, const float* __restrict__ dpp,
    _Float16* __restrict__ Qh, _Float16* __restrict__ Kh,
    _Float16* __restrict__ Vt) {
  const float dw = dwp[0];
  const float dp = dpp[0];
  const size_t N = (size_t)B_ * L_ * H_ * E_;
  for (size_t i = (size_t)blockIdx.x * blockDim.x + threadIdx.x; i < N;
       i += (size_t)gridDim.x * blockDim.x) {
    const int e = (int)(i & 63);
    const int h = (int)((i >> 6) & 7);
    const size_t bl = i >> 9;             // b*L + l
    const int l = (int)(bl & (L_ - 1));
    const int b = (int)(bl >> 11);
    const int bh = b * H_ + h;
    float qq = tanhf(q[i] * inv_q[bh * E_ + e] * dw) * dp;
    float kk = tanhf(k[i] * inv_k[bh * E_ + e] * dw) * dp;
    Qh[((size_t)bh * L_ + l) * E_ + e] = (_Float16)qq;
    Kh[((size_t)bh * S_ + l) * E_ + e] = (_Float16)kk;
    Vt[((size_t)bh * E_ + e) * S_ + l] = (_Float16)v[i];
  }
}

// ---------------------------------------------------------------------------
// Kernel 3: attention. grid = (B*H, L/128), 256 threads = 8 waves.
// K/V superblocks staged cooperatively into LDS (async-to-LDS when available),
// shared by all 8 waves; each wave owns 16 Q rows, two passes over S.
// ---------------------------------------------------------------------------
__global__ __launch_bounds__(256) void attn_kernel(
    const _Float16* __restrict__ Qh, const _Float16* __restrict__ Kh,
    const _Float16* __restrict__ Vt, float* __restrict__ out) {
  __shared__ __align__(16) _Float16 lds_k[128 * KP];      // 18.4 KB
  __shared__ __align__(16) _Float16 lds_v[64 * VP];       // 17.4 KB
  __shared__ __align__(16) _Float16 lds_p[8][16][40];     // 10.2 KB (per-wave P)

  const int bh   = blockIdx.x;
  const int b    = bh >> 3;
  const int h    = bh & 7;
  const int tid  = threadIdx.x;
  const int wave = tid >> 5;
  const int lane = tid & 31;
  const int lh   = lane & 15;
  const int g    = lane >> 4;

  const int m0 = blockIdx.y * 128 + wave * 16;

  const _Float16* Qbh = Qh + (size_t)bh * L_ * E_;
  const _Float16* Kbh = Kh + (size_t)bh * S_ * E_;
  const _Float16* Vbh = Vt + (size_t)bh * E_ * S_;

  // ---- A fragments for this wave's 16 Q rows (row = m0+lh; halves by g) ----
  Frag a0, a1;
  {
    const _Float16* qr = Qbh + (size_t)(m0 + lh) * E_;
    a0.h[0] = *(const v8h*)(qr + g * 16);
    a0.h[1] = *(const v8h*)(qr + g * 16 + 8);
    a1.h[0] = *(const v8h*)(qr + 32 + g * 16);
    a1.h[1] = *(const v8h*)(qr + 32 + g * 16 + 8);
  }

  // ================= Pass 1: row stats (sum, sumsq, max) over S =============
  float sum[8], ssq[8], mx[8];
#pragma unroll
  for (int r = 0; r < 8; ++r) { sum[r] = 0.f; ssq[r] = 0.f; mx[r] = -3.0e30f; }

  for (int n0 = 0; n0 < S_; n0 += 128) {
    // stage K rows [n0, n0+128) x 64 halves: 1024 x 16B chunks over 256 threads
#pragma unroll
    for (int j = 0; j < 4; ++j) {
      const int c = tid + 256 * j;
      const int row = c >> 3, col = c & 7;
      stage16(Kbh + (size_t)(n0 + row) * E_ + col * 8, lds_k + row * KP + col * 8);
    }
    stage_wait();
    __syncthreads();

#pragma unroll
    for (int nt = 0; nt < 8; ++nt) {
      Frag b0, b1;
      const _Float16* kr = lds_k + (nt * 16 + lh) * KP;
      b0.h[0] = *(const v8h*)(kr + g * 16);
      b0.h[1] = *(const v8h*)(kr + g * 16 + 8);
      b1.h[0] = *(const v8h*)(kr + 32 + g * 16);
      b1.h[1] = *(const v8h*)(kr + 32 + g * 16 + 8);
      v8f c = {};
      c = __builtin_amdgcn_wmma_f32_16x16x32_f16(false, a0.v, false, b0.v,
                                                 (short)0, c, false, false);
      c = __builtin_amdgcn_wmma_f32_16x16x32_f16(false, a1.v, false, b1.v,
                                                 (short)0, c, false, false);
#pragma unroll
      for (int r = 0; r < 8; ++r) {
        float x = c[r];
        sum[r] += x; ssq[r] += x * x; mx[r] = fmaxf(mx[r], x);
      }
    }
    __syncthreads();
  }
  // butterfly reduction within each 16-lane half (rows r+8g live there)
#pragma unroll
  for (int m = 1; m < 16; m <<= 1) {
#pragma unroll
    for (int r = 0; r < 8; ++r) {
      sum[r] += __shfl_xor(sum[r], m, 32);
      ssq[r] += __shfl_xor(ssq[r], m, 32);
      mx[r]  = fmaxf(mx[r], __shfl_xor(mx[r], m, 32));
    }
  }

  const float scale = 0.125f;  // 1/sqrt(E)
  float rs[8], adj[8];
#pragma unroll
  for (int r = 0; r < 8; ++r) {
    float var = (ssq[r] - sum[r] * sum[r] * (1.0f / S_)) * (1.0f / (S_ - 1));
    float tau = sqrtf(fmaxf(var, 0.f) + EPS);
    rs[r]  = scale / tau;
    adj[r] = mx[r] * rs[r];
  }

  // ================= Pass 2: softmax * V ====================================
  v8f acc[4];
  {
    v8f z = {};
#pragma unroll
    for (int dt = 0; dt < 4; ++dt) acc[dt] = z;
  }
  float den[8];
#pragma unroll
  for (int r = 0; r < 8; ++r) den[r] = 0.f;

  _Float16* pt = &lds_p[wave][0][0];

  for (int s0 = 0; s0 < S_; s0 += 128) {
    // stage K rows [s0, s0+128) and Vt[0..64)[s0, s0+128) into LDS
#pragma unroll
    for (int j = 0; j < 4; ++j) {
      const int c = tid + 256 * j;
      const int row = c >> 3, col = c & 7;
      stage16(Kbh + (size_t)(s0 + row) * E_ + col * 8, lds_k + row * KP + col * 8);
    }
#pragma unroll
    for (int j = 0; j < 4; ++j) {
      const int c = tid + 256 * j;
      const int d = c >> 4, sc = c & 15;
      stage16(Vbh + (size_t)d * S_ + s0 + sc * 8, lds_v + d * VP + sc * 8);
    }
    stage_wait();
    __syncthreads();

#pragma unroll
    for (int scix = 0; scix < 4; ++scix) {   // 32-column chunks
#pragma unroll
      for (int t = 0; t < 2; ++t) {
        const int rbase = scix * 32 + t * 16;
        Frag b0, b1;
        const _Float16* kr = lds_k + (rbase + lh) * KP;
        b0.h[0] = *(const v8h*)(kr + g * 16);
        b0.h[1] = *(const v8h*)(kr + g * 16 + 8);
        b1.h[0] = *(const v8h*)(kr + 32 + g * 16);
        b1.h[1] = *(const v8h*)(kr + 32 + g * 16 + 8);
        v8f c = {};
        c = __builtin_amdgcn_wmma_f32_16x16x32_f16(false, a0.v, false, b0.v,
                                                   (short)0, c, false, false);
        c = __builtin_amdgcn_wmma_f32_16x16x32_f16(false, a1.v, false, b1.v,
                                                   (short)0, c, false, false);
#pragma unroll
        for (int r = 0; r < 8; ++r) {
          float p = __expf(c[r] * rs[r] - adj[r]);
          den[r] += p;
          pt[(r + 8 * g) * 40 + t * 16 + lh] = (_Float16)p;  // C-layout -> LDS
        }
      }
      // wave-local: make the P-tile stores visible to our own ds_load_b128s
      asm volatile("s_wait_dscnt 0x0" ::: "memory");

      Frag ap;  // P as A-operand: row = lh, k-chunks by g
      {
        const _Float16* pr = pt + lh * 40;
        ap.h[0] = *(const v8h*)(pr + g * 16);
        ap.h[1] = *(const v8h*)(pr + g * 16 + 8);
      }
#pragma unroll
      for (int dt = 0; dt < 4; ++dt) {
        Frag bv;  // Vt row d = dt*16+lh, contiguous s-chunk from LDS
        const _Float16* vr = lds_v + (dt * 16 + lh) * VP + scix * 32 + g * 16;
        bv.h[0] = *(const v8h*)(vr);
        bv.h[1] = *(const v8h*)(vr + 8);
        acc[dt] = __builtin_amdgcn_wmma_f32_16x16x32_f16(
            false, ap.v, false, bv.v, (short)0, acc[dt], false, false);
      }
    }
    __syncthreads();
  }

  // reduce denominators across each 16-lane half
#pragma unroll
  for (int m = 1; m < 16; m <<= 1) {
#pragma unroll
    for (int r = 0; r < 8; ++r) den[r] += __shfl_xor(den[r], m, 32);
  }

  // ---- store: out[b][l][h][d], row = m0 + r + 8g, col = dt*16 + lh ----
  float* ob = out + ((size_t)b * L_ * H_ + h) * E_;
#pragma unroll
  for (int r = 0; r < 8; ++r) {
    const int l = m0 + r + 8 * g;
    const float inv_den = 1.0f / den[r];
#pragma unroll
    for (int dt = 0; dt < 4; ++dt) {
      ob[(size_t)l * H_ * E_ + dt * 16 + lh] = acc[dt][r] * inv_den;
    }
  }
}

// ---------------------------------------------------------------------------
extern "C" void kernel_launch(void* const* d_in, const int* in_sizes, int n_in,
                              void* d_out, int out_size, void* d_ws, size_t ws_size,
                              hipStream_t stream) {
  const float* q  = (const float*)d_in[0];
  const float* k  = (const float*)d_in[1];
  const float* v  = (const float*)d_in[2];
  // d_in[3]: attn_mask (bool) -- unused (mask_flag = False)
  const float* dw = (const float*)d_in[4];
  const float* dp = (const float*)d_in[5];
  float* out = (float*)d_out;

  char* ws = (char*)d_ws;
  float* inv_q = (float*)ws;                           // B*H*E floats
  float* inv_k = inv_q + (size_t)B_ * H_ * E_;         // B*H*E floats
  _Float16* Qh = (_Float16*)(ws + 64 * 1024);          // 8 MB
  _Float16* Kh = Qh + (size_t)B_ * H_ * L_ * E_;       // 8 MB
  _Float16* Vt = Kh + (size_t)B_ * H_ * S_ * E_;       // 8 MB
  (void)in_sizes; (void)n_in; (void)out_size; (void)ws_size;

  stats_kernel<<<dim3(B_ * H_), dim3(64), 0, stream>>>(q, k, inv_q, inv_k);
  transform_kernel<<<dim3(1024), dim3(256), 0, stream>>>(q, k, v, inv_q, inv_k,
                                                         dw, dp, Qh, Kh, Vt);
  attn_kernel<<<dim3(B_ * H_, L_ / 128), dim3(256), 0, stream>>>(Qh, Kh, Vt, out);
}
